// H_vmunet_54975581389251
// MI455X (gfx1250) — matrix operand order repaired and checked by reading, not compiled, not run on hardware
//
#include <hip/hip_runtime.h>

typedef __attribute__((ext_vector_type(16))) _Float16 v16h;
typedef __attribute__((ext_vector_type(8)))  _Float16 h8;
typedef __attribute__((ext_vector_type(4)))  _Float16 h4;
typedef __attribute__((ext_vector_type(8)))  float    v8f;

#define LDA 40  // padded LDS row stride (halves); 80B rows keep 16B alignment

// ---------------------------------------------------------------------------
// Batched WMMA GEMM:  C[z] = W[z] (MxK) * X[z] (KxN) [+ bias[z]]   row-major.
// Macro tile 32(M) x 64(N), BK=32. 128 threads = 4 waves; each wave owns a
// 16x32 strip: one A fragment, two B fragments, two f32 accumulators ->
// 2 x v_wmma_f32_16x16x32_f16 per K-chunk. A kept [m][k] in LDS, B kept
// transposed [n][k] so all fragment gathers are contiguous ds_load_b128.
// Main loop handles only full 32-wide K chunks (pure b128 traffic, branchless
// M-edge via clamped row + 0/1 scale); one tail chunk handles ragged K with
// clamped always-valid addresses (selects, no exec-mask load serialization).
// ---------------------------------------------------------------------------
__global__ void __launch_bounds__(128) gemm_wmma_kernel(
    const float* __restrict__ W, const float* __restrict__ X,
    const float* __restrict__ bias, float* __restrict__ C,
    int M, int N, int K,
    long long wStride, long long xStride, long long cStride, int biasStride)
{
  __shared__ _Float16 As[32 * LDA];   // [m][k]
  __shared__ _Float16 Bt[64 * LDA];   // [n][k] (transposed)

  const int bz = blockIdx.z;
  const float* Wb = W + (long long)bz * wStride;
  const float* Xb = X + (long long)bz * xStride;
  float*       Cb = C + (long long)bz * cStride;

  const int m0   = blockIdx.y * 32;
  const int n0   = blockIdx.x * 64;
  const int tid  = threadIdx.x;
  const int lane = tid & 31;
  const int wv   = tid >> 5;
  const int half = lane >> 4;          // 0: lanes 0-15, 1: lanes 16-31
  const int row  = lane & 15;          // M index for A-frag, N index for B-frag
  const int wm   = (wv >> 1) << 4;     // 0 or 16
  const int wn   = (wv & 1) << 5;      // 0 or 32

  // fill coordinates
  const int a_mi = tid >> 2;           // 0..31
  const int a_kq = (tid & 3) * 8;      // 0,8,16,24
  const int b_kq = (tid & 7) * 4;      // 0..28
  const int b_nq = (tid >> 3) * 4;     // 0..60

  v8f acc0 = {0.f, 0.f, 0.f, 0.f, 0.f, 0.f, 0.f, 0.f};
  v8f acc1 = {0.f, 0.f, 0.f, 0.f, 0.f, 0.f, 0.f, 0.f};

  // A-row handling (loop invariant): clamped row + 0/1 scale, branchless
  const int  gm   = m0 + a_mi;
  const int  gmc  = (gm < M) ? gm : (M - 1);
  const float sA  = (gm < M) ? 1.f : 0.f;
  const float* Wrow = Wb + (long long)gmc * K;

  auto mma_step = [&]() {
    const _Float16* Ar = As + (wm + row) * LDA + half * 16;
    h8 a0 = *(const h8*)(Ar);
    h8 a1 = *(const h8*)(Ar + 8);
    v16h av = __builtin_shufflevector(a0, a1, 0,1,2,3,4,5,6,7,8,9,10,11,12,13,14,15);

    const _Float16* Br0 = Bt + (wn + row) * LDA + half * 16;
    h8 b00 = *(const h8*)(Br0);
    h8 b01 = *(const h8*)(Br0 + 8);
    v16h bv0 = __builtin_shufflevector(b00, b01, 0,1,2,3,4,5,6,7,8,9,10,11,12,13,14,15);

    const _Float16* Br1 = Bt + (wn + 16 + row) * LDA + half * 16;
    h8 b10 = *(const h8*)(Br1);
    h8 b11 = *(const h8*)(Br1 + 8);
    v16h bv1 = __builtin_shufflevector(b10, b11, 0,1,2,3,4,5,6,7,8,9,10,11,12,13,14,15);

    acc0 = __builtin_amdgcn_wmma_f32_16x16x32_f16(false, av, false, bv0,
                                                  (short)0, acc0, false, false);
    acc1 = __builtin_amdgcn_wmma_f32_16x16x32_f16(false, av, false, bv1,
                                                  (short)0, acc1, false, false);
  };

  const bool kvec  = ((K & 3) == 0);
  const int  Kfull = kvec ? (K & ~31) : 0;

  int k0 = 0;
  for (; k0 < Kfull; k0 += 32) {
    // ---- A tile: two global_load_b128, branchless edge
    {
      const float4 f0 = *(const float4*)(Wrow + k0 + a_kq);
      const float4 f1 = *(const float4*)(Wrow + k0 + a_kq + 4);
      h8 hv;
      hv[0] = (_Float16)(f0.x * sA); hv[1] = (_Float16)(f0.y * sA);
      hv[2] = (_Float16)(f0.z * sA); hv[3] = (_Float16)(f0.w * sA);
      hv[4] = (_Float16)(f1.x * sA); hv[5] = (_Float16)(f1.y * sA);
      hv[6] = (_Float16)(f1.z * sA); hv[7] = (_Float16)(f1.w * sA);
      *(h8*)(As + a_mi * LDA + a_kq) = hv;   // ds_store_b128
    }
    // ---- B tile: 4 global_load_b128, store transposed (ds_store_b64 x4)
    {
      _Float16 hv[4][4];
#pragma unroll
      for (int j = 0; j < 4; ++j) {
        const float4 f = *(const float4*)(Xb + (long long)(k0 + b_kq + j) * N + n0 + b_nq);
        hv[0][j] = (_Float16)f.x; hv[1][j] = (_Float16)f.y;
        hv[2][j] = (_Float16)f.z; hv[3][j] = (_Float16)f.w;
      }
#pragma unroll
      for (int i = 0; i < 4; ++i) {
        h4 t;
#pragma unroll
        for (int j = 0; j < 4; ++j) t[j] = hv[i][j];
        *(h4*)(Bt + (b_nq + i) * LDA + b_kq) = t;
      }
    }
    // prefetch next K-chunk rows (global_prefetch_b8)
    if (k0 + 32 < Kfull) {
      __builtin_prefetch(Wrow + k0 + 32 + a_kq, 0, 1);
      __builtin_prefetch(Xb + (long long)(k0 + 32 + b_kq) * N + n0 + b_nq, 0, 1);
    }
    __syncthreads();
    mma_step();
    __syncthreads();
  }

  if (k0 < K) {
    // ---- tail chunk: clamped always-valid addresses, pure selects
    {
      h8 hv;
#pragma unroll
      for (int e = 0; e < 8; ++e) {
        const int gk  = k0 + a_kq + e;
        const int gkc = (gk < K) ? gk : (K - 1);
        const float v = Wrow[gkc];
        hv[e] = (_Float16)((gk < K) ? v * sA : 0.f);
      }
      *(h8*)(As + a_mi * LDA + a_kq) = hv;
    }
    {
      _Float16 hv[4][4];
#pragma unroll
      for (int j = 0; j < 4; ++j) {
        const int gk  = k0 + b_kq + j;
        const int gkc = (gk < K) ? gk : (K - 1);
        const float s = (gk < K) ? 1.f : 0.f;
        const float4 f = *(const float4*)(Xb + (long long)gkc * N + n0 + b_nq);
        hv[0][j] = (_Float16)(f.x * s); hv[1][j] = (_Float16)(f.y * s);
        hv[2][j] = (_Float16)(f.z * s); hv[3][j] = (_Float16)(f.w * s);
      }
#pragma unroll
      for (int i = 0; i < 4; ++i) {
        h4 t;
#pragma unroll
        for (int j = 0; j < 4; ++j) t[j] = hv[i][j];
        *(h4*)(Bt + (b_nq + i) * LDA + b_kq) = t;
      }
    }
    __syncthreads();
    mma_step();
  }

  // D layout: VGPR r -> M = r + 8*half ; N = lane&15
#pragma unroll
  for (int r = 0; r < 8; ++r) {
    const int m = m0 + wm + r + half * 8;
    const int n = n0 + wn + row;
    if (m < M) {
      float bval = bias ? bias[(long long)biasStride * bz + m] : 0.f;
      if (n < N)        Cb[(long long)m * N + n]      = acc0[r] + bval;
      if ((n + 16) < N) Cb[(long long)m * N + n + 16] = acc1[r] + bval;
    }
  }
}

// ---------------------------------------------------------------------------
// Depthwise conv (ks=7 pad=3 or ks=3 pad=1), optional SiLU.
// ---------------------------------------------------------------------------
__global__ void dwconv_kernel(const float* __restrict__ x, long long xStride,
                              const float* __restrict__ w,
                              const float* __restrict__ bias,
                              float* __restrict__ y, long long yStride,
                              int B, int C, int H, int Wd, int ks, int pad, int act)
{
  long long idx = (long long)blockIdx.x * blockDim.x + threadIdx.x;
  long long total = (long long)B * C * H * Wd;
  if (idx >= total) return;
  int wi = (int)(idx % Wd);
  int hi = (int)((idx / Wd) % H);
  int c  = (int)((idx / ((long long)H * Wd)) % C);
  int b  = (int)(idx / ((long long)C * H * Wd));
  const float* xb = x + (long long)b * xStride + (long long)c * H * Wd;
  const float* wc = w + (long long)c * ks * ks;
  float acc = bias ? bias[c] : 0.f;
  for (int r = 0; r < ks; ++r) {
    int yy = hi + r - pad;
    if (yy < 0 || yy >= H) continue;
    for (int s = 0; s < ks; ++s) {
      int xx = wi + s - pad;
      if (xx < 0 || xx >= Wd) continue;
      acc += wc[r * ks + s] * xb[yy * Wd + xx];
    }
  }
  if (act) acc = acc / (1.f + __expf(-acc));   // SiLU
  y[(long long)b * yStride + (long long)c * H * Wd + (long long)hi * Wd + wi] = acc;
}

// 4-direction stack: [x1, transposeHW(x1), flip(x1), flip(transposeHW(x1))]
__global__ void build_xs_kernel(const float* __restrict__ xc,
                                float* __restrict__ xs, int d, int H, int Wd)
{
  int L = H * Wd;
  long long idx = (long long)blockIdx.x * blockDim.x + threadIdx.x;
  if (idx >= (long long)d * L) return;
  int l  = (int)(idx % L);
  int dd = (int)(idx / L);
  int hh = l / Wd, ww = l % Wd;
  int lt = ww * H + hh;
  float v = xc[idx];
  long long dL = (long long)d * L;
  long long base = (long long)dd * L;
  xs[base + l]                   = v;
  xs[dL + base + lt]             = v;
  xs[2 * dL + base + (L - 1 - l)]  = v;
  xs[3 * dL + base + (L - 1 - lt)] = v;
}

// Selective scan: one thread per (k, d). dt_b already added by GEMM bias.
__global__ void scan_kernel(const float* __restrict__ xs,
                            const float* __restrict__ xdbl,
                            const float* __restrict__ dts,
                            const float* __restrict__ A_log,
                            const float* __restrict__ Dp,
                            float* __restrict__ ys,
                            int d, int L, int Crows, int R)
{
  int idx = blockIdx.x * blockDim.x + threadIdx.x;
  if (idx >= 4 * d) return;
  int k = idx / d, dd = idx % d;
  const float* u  = xs   + ((long long)k * d + dd) * L;
  const float* dr = dts  + ((long long)k * d + dd) * L;
  const float* Bv = xdbl + ((long long)k * Crows + R) * L;        // B rows
  const float* Cv = xdbl + ((long long)k * Crows + R + 16) * L;   // C rows
  float*       yo = ys   + ((long long)k * d + dd) * L;
  float Ad[16], h[16];
#pragma unroll
  for (int n = 0; n < 16; ++n) {
    Ad[n] = -__expf(A_log[((long long)k * d + dd) * 16 + n]);
    h[n]  = 0.f;
  }
  float Dk = Dp[k * d + dd];
  for (int l = 0; l < L; ++l) {
    float dt = dr[l];
    dt = (dt > 20.f) ? dt : log1pf(__expf(dt));   // softplus
    float ut = u[l];
    float du = dt * ut;
    float y = 0.f;
#pragma unroll
    for (int n = 0; n < 16; ++n) {
      float hn = __expf(dt * Ad[n]) * h[n] + du * Bv[(long long)n * L + l];
      h[n] = hn;
      y += hn * Cv[(long long)n * L + l];
    }
    yo[l] = y + ut * Dk;
  }
}

// y = ys0 + flip(ys2) + unwh(ys1) + unwh(flip(ys3))
__global__ void combine_kernel(const float* __restrict__ ys,
                               float* __restrict__ yc, int d, int H, int Wd)
{
  int L = H * Wd;
  long long idx = (long long)blockIdx.x * blockDim.x + threadIdx.x;
  if (idx >= (long long)d * L) return;
  int l  = (int)(idx % L);
  int dd = (int)(idx / L);
  int hh = l / Wd, ww = l % Wd;
  int lt = ww * H + hh;
  long long dL = (long long)d * L;
  long long base = (long long)dd * L;
  yc[idx] = ys[base + l]
          + ys[2 * dL + base + (L - 1 - l)]
          + ys[dL + base + lt]
          + ys[3 * dL + base + (L - 1 - lt)];
}

// LayerNorm over channels (stride-L in NCHW) then * silu(z)
__global__ void ln_gate_kernel(const float* __restrict__ yc,
                               const float* __restrict__ z,
                               const float* __restrict__ g,
                               const float* __restrict__ bb,
                               float* __restrict__ yg, int d, int L)
{
  int l = blockIdx.x * blockDim.x + threadIdx.x;
  if (l >= L) return;
  float m = 0.f;
  for (int dd = 0; dd < d; ++dd) m += yc[(long long)dd * L + l];
  m /= (float)d;
  float v = 0.f;
  for (int dd = 0; dd < d; ++dd) {
    float t = yc[(long long)dd * L + l] - m;
    v += t * t;
  }
  v /= (float)d;
  float rs = rsqrtf(v + 1e-5f);
  for (int dd = 0; dd < d; ++dd) {
    float val = (yc[(long long)dd * L + l] - m) * rs * g[dd] + bb[dd];
    float zz = z[(long long)dd * L + l];
    yg[(long long)dd * L + l] = val * (zz / (1.f + __expf(-zz)));
  }
}

// o[z] = a[z] * b[z] with per-batch strides
__global__ void mul_kernel(const float* __restrict__ a, long long as,
                           const float* __restrict__ b, long long bs,
                           float* __restrict__ o, long long os,
                           long long per, int nb)
{
  long long idx = (long long)blockIdx.x * blockDim.x + threadIdx.x;
  if (idx >= per * nb) return;
  long long bt = idx / per, r = idx % per;
  o[bt * os + r] = a[bt * as + r] * b[bt * bs + r];
}

// ---------------------------------------------------------------------------
extern "C" void kernel_launch(void* const* d_in, const int* in_sizes, int n_in,
                              void* d_out, int out_size, void* d_ws, size_t ws_size,
                              hipStream_t stream) {
  (void)in_sizes; (void)n_in; (void)out_size; (void)ws_size;
  const int B = 4, H = 64, Wd = 64, L = H * Wd;       // L = 4096
  const int DIMs[5] = {8, 16, 32, 64, 128};
  const int offs[6] = {0, 8, 24, 56, 120, 248};

  auto IN = [&](int i) { return (const float*)d_in[i]; };
  const float* Xin    = IN(0);
  const float* pin_w  = IN(1);
  const float* pin_b  = IN(2);
  const float* dw_w   = IN(3);
  const float* dw_b   = IN(4);
  const float* pout_w = IN(5);
  const float* pout_b = IN(6);
  const float* pw_w[4], *pw_b[4];
  for (int i = 0; i < 4; ++i) { pw_w[i] = IN(7 + 2 * i); pw_b[i] = IN(8 + 2 * i); }

  // workspace layout (floats); sized for the largest stage (di = 256)
  float* ws = (float*)d_ws;
  size_t off = 0;
  auto alloc = [&](size_t n) { float* p = ws + off; off += n; return p; };
  float* fused = alloc((size_t)B * 256 * L);
  float* dwb   = alloc((size_t)B * 248 * L);
  float* hbuf  = alloc((size_t)B * 128 * L);
  float* h2buf = alloc((size_t)B * 128 * L);
  float* xz    = alloc((size_t)512 * L);
  float* xc    = alloc((size_t)256 * L);
  float* xsb   = alloc((size_t)4 * 256 * L);
  float* xdbl  = alloc((size_t)4 * 40 * L);
  float* dtb   = alloc((size_t)4 * 256 * L);
  float* ysb   = alloc((size_t)4 * 256 * L);
  float* ycb   = alloc((size_t)256 * L);
  float* ygb   = alloc((size_t)256 * L);

  auto gemm = [&](const float* Wp, const float* Xp, const float* bp, float* Cp,
                  int M, int N, int K, long long wS, long long xS, long long cS,
                  int bStr, int nb) {
    dim3 g((N + 63) / 64, (M + 31) / 32, nb);
    gemm_wmma_kernel<<<g, 128, 0, stream>>>(Wp, Xp, bp, Cp, M, N, K, wS, xS, cS, bStr);
  };

  // 1) fused = conv1x1(x, proj_in) : (B,256,L)
  gemm(pin_w, Xin, pin_b, fused, 256, L, 128, 0, (long long)128 * L,
       (long long)256 * L, 0, B);

  // 2) dw_abc = dwconv7x7(fused[:,8:256]) : (B,248,L)
  {
    long long tot = (long long)B * 248 * L;
    dwconv_kernel<<<dim3((unsigned)((tot + 255) / 256)), 256, 0, stream>>>(
        fused + 8LL * L, (long long)256 * L, dw_w, dw_b,
        dwb, (long long)248 * L, B, 248, H, Wd, 7, 3, 0);
  }

  // 3) h = pwa * dw_list[0] : (B,8,L)
  {
    long long per = 8LL * L;
    mul_kernel<<<dim3((unsigned)((per * B + 255) / 256)), 256, 0, stream>>>(
        fused, (long long)256 * L, dwb, (long long)248 * L, hbuf, per, per, B);
  }

  // 4) stage chain: ss2d_0, (pw_i -> *dw_list[i+1] -> ss2d_{i+1})*
  for (int st = 0; st < 5; ++st) {
    int dm = DIMs[st], di = 2 * dm;
    int R  = (dm + 15) / 16; if (R < 1) R = 1;
    int Cr = R + 32;
    int base = 15 + 11 * st;
    const float* ipw  = IN(base + 0);
    const float* cw   = IN(base + 1);
    const float* cb   = IN(base + 2);
    const float* xpw  = IN(base + 3);
    const float* dtw  = IN(base + 4);
    const float* dtbp = IN(base + 5);
    const float* Alog = IN(base + 6);
    const float* Dp   = IN(base + 7);
    const float* lng  = IN(base + 8);
    const float* lnb  = IN(base + 9);
    const float* opw  = IN(base + 10);

    for (int b = 0; b < B; ++b) {
      float* hin = hbuf + (long long)b * dm * L;
      // xz = in_proj (2di x L)
      gemm(ipw, hin, nullptr, xz, 2 * di, L, dm, 0, 0, 0, 0, 1);
      // xc = silu(dwconv3x3(xz[:di]))
      {
        long long tot = (long long)di * L;
        dwconv_kernel<<<dim3((unsigned)((tot + 255) / 256)), 256, 0, stream>>>(
            xz, 0, cw, cb, xc, 0, 1, di, H, Wd, 3, 1, 1);
      }
      // xs = 4-direction stack
      {
        long long tot = (long long)di * L;
        build_xs_kernel<<<dim3((unsigned)((tot + 255) / 256)), 256, 0, stream>>>(
            xc, xsb, di, H, Wd);
      }
      // x_dbl[k] = x_proj_w[k] @ xs[k]   (Cr x L, batched over k=0..3)
      gemm(xpw, xsb, nullptr, xdbl, Cr, L, di,
           (long long)Cr * di, (long long)di * L, (long long)Cr * L, 0, 4);
      // dt_raw[k] = dt_w[k] @ dts[k] + dt_b[k]   (di x L, K = R)
      gemm(dtw, xdbl, dtbp, dtb, di, L, R,
           (long long)di * R, (long long)Cr * L, (long long)di * L, di, 4);
      // selective scan
      {
        int tot = 4 * di;
        scan_kernel<<<dim3((unsigned)((tot + 63) / 64)), 64, 0, stream>>>(
            xsb, xdbl, dtb, Alog, Dp, ysb, di, L, Cr, R);
      }
      // combine 4 directions
      {
        long long tot = (long long)di * L;
        combine_kernel<<<dim3((unsigned)((tot + 255) / 256)), 256, 0, stream>>>(
            ysb, ycb, di, H, Wd);
      }
      // layernorm + silu gate (z = xz[di:2di])
      ln_gate_kernel<<<dim3((unsigned)((L + 255) / 256)), 256, 0, stream>>>(
          ycb, xz + (long long)di * L, lng, lnb, ygb, di, L);
      // h = out_proj (dm x L) written back in place
      gemm(opw, ygb, nullptr, hin, dm, L, di, 0, 0, 0, 0, 1);
    }

    if (st < 4) {
      int Ci = DIMs[st], Co = DIMs[st + 1];
      // h2 = conv1x1(h, pw_st) : (B,Co,L)
      gemm(pw_w[st], hbuf, pw_b[st], h2buf, Co, L, Ci,
           0, (long long)Ci * L, (long long)Co * L, 0, B);
      // h = h2 * dw_list[st+1]
      long long per = (long long)Co * L;
      mul_kernel<<<dim3((unsigned)((per * B + 255) / 256)), 256, 0, stream>>>(
          h2buf, per, dwb + (long long)offs[st + 1] * L, (long long)248 * L,
          hbuf, per, per, B);
    }
  }

  // 5) out = conv1x1(h, proj_out) : (B,128,L)  -> d_out (fully overwritten)
  gemm(pout_w, hbuf, pout_b, (float*)d_out, 128, L, 128,
       0, (long long)128 * L, (long long)128 * L, 0, B);
}
